// Attention_49409303773358
// MI455X (gfx1250) — matrix operand order repaired and checked by reading, not compiled, not run on hardware
//
#include <hip/hip_runtime.h>

typedef __attribute__((ext_vector_type(16))) __bf16 bf16x16;
typedef __attribute__((ext_vector_type(8)))  __bf16 bf16x8;
typedef __attribute__((ext_vector_type(8)))  float  f32x8;

// Problem constants (fixed by setup_inputs)
#define BATCH 4
#define NTOK  1025
#define DIMX  768
#define HEADS 12
#define DH    64
#define QKVD  2304        // 3*768
#define MROWS 4100        // BATCH*NTOK
#define MMROWS 4096       // BATCH*(NTOK-1)

__device__ __forceinline__ f32x8 wmma_bf16(bf16x16 a, bf16x16 b, f32x8 c) {
  return __builtin_amdgcn_wmma_f32_16x16x32_bf16(false, a, false, b, (short)0, c, false, false);
}

// A fragment: 16x32 bf16 tile at (row0, k0) from LDS, row-major, stride elems.
// ISA layout: lane<16 -> row=lane, K {k0..k0+7, k0+16..k0+23}; lane>=16 -> +8 K offset.
__device__ __forceinline__ bf16x16 ldsA(const __bf16* base, int row0, int stride, int k0, int lane) {
  int half = (lane >> 4) & 1, r = lane & 15;
  const __bf16* p = base + (row0 + r) * stride + k0 + half * 8;
  bf16x8 lo = *(const bf16x8*)p;
  bf16x8 hi = *(const bf16x8*)(p + 16);
  bf16x16 a;
#pragma unroll
  for (int i = 0; i < 8; ++i) { a[i] = lo[i]; a[i + 8] = hi[i]; }
  return a;
}

// B fragment: K x 16 tile; LDS holds B as [N][K] row-major (K contiguous per column).
// lane<16 -> col=lane, K {k0..k0+7, k0+16..k0+23}; lane>=16 -> +8 K offset.
__device__ __forceinline__ bf16x16 ldsB(const __bf16* base, int n0, int stride, int k0, int lane) {
  int half = (lane >> 4) & 1, r = lane & 15;
  const __bf16* p = base + (n0 + r) * stride + k0 + half * 8;
  bf16x8 lo = *(const bf16x8*)p;
  bf16x8 hi = *(const bf16x8*)(p + 16);
  bf16x16 b;
#pragma unroll
  for (int i = 0; i < 8; ++i) { b[i] = lo[i]; b[i + 8] = hi[i]; }
  return b;
}

// ---------------------------------------------------------------------------
// Generic tiled GEMM: OUT[M,N] = rowscale(X)[M,K] * W[N,K]^T (+bias)
// Workgroup tile 128x64: 4 waves, each wave 32x64 (2 A-frags x 4 B-frags).
// outMode 0: OUT[row*N+col]
// outMode 1: row = b*1025+n ; skip n==0 ; OUT[(b*1024+n-1)*768+col]  (m output)
// ---------------------------------------------------------------------------
__global__ void __launch_bounds__(128)
gemm_wmma(const float* __restrict__ X, const float* __restrict__ W,
          const float* __restrict__ bias, const float* __restrict__ rowscale,
          float* __restrict__ OUT, int M, int N, int K, int absW, int outMode) {
  __shared__ __attribute__((aligned(16))) __bf16 sX[128 * 32];
  __shared__ __attribute__((aligned(16))) __bf16 sW[64 * 32];
  const int tid = threadIdx.x, wave = tid >> 5, lane = tid & 31;
  const int m0 = blockIdx.x * 128, n0 = blockIdx.y * 64;

  f32x8 acc[2][4];
#pragma unroll
  for (int g = 0; g < 2; ++g)
#pragma unroll
    for (int n = 0; n < 4; ++n)
#pragma unroll
      for (int r = 0; r < 8; ++r) acc[g][n][r] = 0.0f;

  for (int kt = 0; kt < K; kt += 32) {
#pragma unroll 4
    for (int e = tid; e < 128 * 32; e += 128) {
      int r = e >> 5, c = e & 31;
      int row = m0 + r; row = row < M ? row : (M - 1);
      float v = X[(size_t)row * K + kt + c];
      if (rowscale) v *= rowscale[row];
      sX[e] = (__bf16)v;
    }
#pragma unroll 4
    for (int e = tid; e < 64 * 32; e += 128) {
      int r = e >> 5, c = e & 31;
      float v = W[(size_t)(n0 + r) * K + kt + c];
      if (absW) v = fabsf(v);
      sW[e] = (__bf16)v;
    }
    __syncthreads();
    bf16x16 a0 = ldsA(sX, wave * 32, 32, 0, lane);
    bf16x16 a1 = ldsA(sX, wave * 32 + 16, 32, 0, lane);
#pragma unroll
    for (int n = 0; n < 4; ++n) {
      bf16x16 b = ldsB(sW, n * 16, 32, 0, lane);
      acc[0][n] = wmma_bf16(a0, b, acc[0][n]);
      acc[1][n] = wmma_bf16(a1, b, acc[1][n]);
    }
    __syncthreads();
  }

  const int half = (lane >> 4) & 1, r15 = lane & 15;
#pragma unroll
  for (int g = 0; g < 2; ++g) {
#pragma unroll
    for (int n = 0; n < 4; ++n) {
      int col = n0 + n * 16 + r15;
#pragma unroll
      for (int r = 0; r < 8; ++r) {
        int row = m0 + wave * 32 + g * 16 + half * 8 + r;
        if (row < M) {
          float v = acc[g][n][r];
          if (bias) v += bias[col];
          if (outMode == 0) {
            OUT[(size_t)row * N + col] = v;
          } else {
            int b = row / NTOK, nn = row % NTOK;
            if (nn >= 1) OUT[((size_t)b * (NTOK - 1) + (nn - 1)) * DIMX + col] = v;
          }
        }
      }
    }
  }
}

// ---------------------------------------------------------------------------
// colmax[b*2304+c] = max over 1024 patch tokens of qkvm[b, n, c]
// ---------------------------------------------------------------------------
__global__ void colmax_kernel(const float* __restrict__ qkvm, float* __restrict__ colmax) {
  int c = blockIdx.x * blockDim.x + threadIdx.x;
  if (c >= BATCH * QKVD) return;
  int b = c / QKVD, col = c % QKVD;
  const float* p = qkvm + (size_t)b * (NTOK - 1) * QKVD + col;
  float mx = 0.0f;  // entries are mask @ |W|^T >= 0
  for (int n = 0; n < NTOK - 1; ++n) mx = fmaxf(mx, p[(size_t)n * QKVD]);
  colmax[c] = mx;
}

// updated[b*1025+n] : 1 for CLS, else mean(mask[b,n-1,:]) > 0
__global__ void updated_kernel(const float* __restrict__ mask, float* __restrict__ upd) {
  int row = blockIdx.x * blockDim.x + threadIdx.x;
  if (row >= MROWS) return;
  int b = row / NTOK, n = row % NTOK;
  float u = 1.0f;
  if (n >= 1) {
    const float* p = mask + ((size_t)b * (NTOK - 1) + (n - 1)) * DIMX;
    float s = 0.0f;
    for (int i = 0; i < DIMX; ++i) s += p[i];
    u = (s > 0.0f) ? 1.0f : 0.0f;
  }
  upd[row] = u;
}

// normalized mask value for token tok (1..1024), column col of qkvm
__device__ __forceinline__ float norm_m(const float* __restrict__ qkvm,
                                        const float* __restrict__ colmax,
                                        int b, int tok, int col) {
  if (tok == 0) return 1.0f;
  float v = qkvm[((size_t)b * (NTOK - 1) + (tok - 1)) * QKVD + col];
  return v / (1e-6f + colmax[b * QKVD + col]);
}

// ---------------------------------------------------------------------------
// Flash attention with dual accumulators:
//   outh = softmax(q'k'^T * scale) @ (v * v_m)
//   mh   = softmax(...) @ v_m
// grid = (B*H, 17), block = 128 (4 waves, 16 query rows each)
// ---------------------------------------------------------------------------
__global__ void __launch_bounds__(128)
attn_kernel(const float* __restrict__ qkv, const float* __restrict__ qkvm,
            const float* __restrict__ colmax, float* __restrict__ outh,
            float* __restrict__ mh) {
  const int b = blockIdx.x / HEADS, h = blockIdx.x % HEADS;
  const int wave = threadIdx.x >> 5, lane = threadIdx.x & 31;
  const int tid = threadIdx.x;
  const int row0 = (blockIdx.y * 4 + wave) * 16;  // first query token of this wave
  const float scale = 0.125f;                      // 64^-0.5

  __shared__ __attribute__((aligned(16))) __bf16 sQ[4][16 * 64];
  __shared__ __attribute__((aligned(16))) __bf16 sK[32 * 64];   // [key][d]
  __shared__ __attribute__((aligned(16))) __bf16 sVo[64 * 32];  // [d][key]  v*v_m
  __shared__ __attribute__((aligned(16))) __bf16 sVm[64 * 32];  // [d][key]  v_m
  __shared__ __attribute__((aligned(16))) __bf16 sP[4][16 * 32];

  // Stage q' = q * q_m for this wave's 16 rows
  for (int e = lane; e < 16 * 64; e += 32) {
    int r = e >> 6, d = e & 63;
    int tok = row0 + r;
    float v = 0.0f;
    if (tok < NTOK) {
      int col = h * DH + d;
      v = qkv[((size_t)b * NTOK + tok) * QKVD + col] * norm_m(qkvm, colmax, b, tok, col);
    }
    sQ[wave][e] = (__bf16)v;
  }
  bf16x16 qA0 = ldsA(sQ[wave], 0, 64, 0, lane);
  bf16x16 qA1 = ldsA(sQ[wave], 0, 64, 32, lane);

  f32x8 accO[4], accM[4];
#pragma unroll
  for (int n = 0; n < 4; ++n)
#pragma unroll
    for (int r = 0; r < 8; ++r) { accO[n][r] = 0.0f; accM[n][r] = 0.0f; }
  float mrow[8], lrow[8];
#pragma unroll
  for (int r = 0; r < 8; ++r) { mrow[r] = -3.0e38f; lrow[r] = 0.0f; }

  const int half = (lane >> 4) & 1, r15 = lane & 15;

  for (int kb = 0; kb < 33; ++kb) {       // 33*32 = 1056 >= 1025 keys
    const int kb0 = kb * 32;
    // stage K', Vo, Vm (whole workgroup)
    for (int e = tid; e < 32 * 64; e += 128) {
      int key = e >> 6, d = e & 63;
      int tok = kb0 + key;
      float kk = 0.0f, vo = 0.0f, vm = 0.0f;
      if (tok < NTOK) {
        int ck = DIMX + h * DH + d, cv = 2 * DIMX + h * DH + d;
        size_t base = ((size_t)b * NTOK + tok) * QKVD;
        float kmn = norm_m(qkvm, colmax, b, tok, ck);
        float vmn = norm_m(qkvm, colmax, b, tok, cv);
        kk = qkv[base + ck] * kmn;
        vm = vmn;
        vo = qkv[base + cv] * vmn;
      }
      sK[key * 64 + d] = (__bf16)kk;
      sVo[d * 32 + key] = (__bf16)vo;
      sVm[d * 32 + key] = (__bf16)vm;
    }
    __syncthreads();

    // S = q' k'^T * scale  (two 16x16 key sub-tiles)
    f32x8 S[2];
#pragma unroll
    for (int st = 0; st < 2; ++st) {
      f32x8 s;
#pragma unroll
      for (int r = 0; r < 8; ++r) s[r] = 0.0f;
      s = wmma_bf16(qA0, ldsB(sK, st * 16, 64, 0, lane), s);
      s = wmma_bf16(qA1, ldsB(sK, st * 16, 64, 32, lane), s);
      int keycol = kb0 + st * 16 + r15;
      float msk = (keycol < NTOK) ? 0.0f : -3.0e38f;
#pragma unroll
      for (int r = 0; r < 8; ++r) s[r] = s[r] * scale + msk;
      S[st] = s;
    }

    // online softmax row statistics (rows live across 16-lane halves)
    float pv[2][8];
#pragma unroll
    for (int r = 0; r < 8; ++r) {
      float v = fmaxf(S[0][r], S[1][r]);
#pragma unroll
      for (int off = 1; off < 16; off <<= 1) v = fmaxf(v, __shfl_xor(v, off, 32));
      float mnew = fmaxf(mrow[r], v);
      float alpha = __expf(mrow[r] - mnew);
      float p0 = __expf(S[0][r] - mnew);
      float p1 = __expf(S[1][r] - mnew);
      pv[0][r] = p0; pv[1][r] = p1;
      float rs = p0 + p1;
#pragma unroll
      for (int off = 1; off < 16; off <<= 1) rs += __shfl_xor(rs, off, 32);
      lrow[r] = lrow[r] * alpha + rs;
      mrow[r] = mnew;
#pragma unroll
      for (int n = 0; n < 4; ++n) { accO[n][r] *= alpha; accM[n][r] *= alpha; }
    }

    // P (C-layout) -> per-wave LDS -> A-layout fragment
#pragma unroll
    for (int st = 0; st < 2; ++st)
#pragma unroll
      for (int r = 0; r < 8; ++r)
        sP[wave][(half * 8 + r) * 32 + st * 16 + r15] = (__bf16)pv[st][r];

    bf16x16 pA = ldsA(sP[wave], 0, 32, 0, lane);
#pragma unroll
    for (int n = 0; n < 4; ++n) {
      accO[n] = wmma_bf16(pA, ldsB(sVo, n * 16, 32, 0, lane), accO[n]);
      accM[n] = wmma_bf16(pA, ldsB(sVm, n * 16, 32, 0, lane), accM[n]);
    }
    __syncthreads();
  }

  float inv[8];
#pragma unroll
  for (int r = 0; r < 8; ++r) inv[r] = 1.0f / lrow[r];
#pragma unroll
  for (int n = 0; n < 4; ++n) {
    int d = n * 16 + r15;
#pragma unroll
    for (int r = 0; r < 8; ++r) {
      int tok = row0 + half * 8 + r;
      if (tok < NTOK) {
        size_t idx = ((size_t)b * NTOK + tok) * DIMX + h * DH + d;
        outh[idx] = accO[n][r] * inv[r];
        mh[idx]   = accM[n][r] * inv[r];
      }
    }
  }
}

// ---------------------------------------------------------------------------
extern "C" void kernel_launch(void* const* d_in, const int* in_sizes, int n_in,
                              void* d_out, int out_size, void* d_ws, size_t ws_size,
                              hipStream_t stream) {
  const float* x     = (const float*)d_in[0];  // [4,1025,768]
  const float* mask  = (const float*)d_in[1];  // [4,1024,768]
  const float* w_qkv = (const float*)d_in[2];  // [2304,768]
  const float* w_out = (const float*)d_in[3];  // [768,768]
  const float* b_out = (const float*)d_in[4];  // [768]
  (void)in_sizes; (void)n_in; (void)out_size; (void)ws_size;

  float* ws = (float*)d_ws;
  float* qkv    = ws;                                  // 4100*2304
  float* qkvm   = qkv  + (size_t)MROWS * QKVD;         // 4096*2304
  float* cmax   = qkvm + (size_t)MMROWS * QKVD;        // 4*2304
  float* upd    = cmax + (size_t)BATCH * QKVD;         // 4100
  float* outh   = upd  + MROWS;                        // 4100*768
  float* mh     = outh + (size_t)MROWS * DIMX;         // 4100*768

  float* out0 = (float*)d_out;                          // out  [4,1025,768]
  float* out1 = out0 + (size_t)MROWS * DIMX;            // m[:,1:] [4,1024,768]

  // qkv = x @ w_qkv^T ; qkv_m = mask @ |w_qkv|^T
  gemm_wmma<<<dim3(33, 36), 128, 0, stream>>>(x, w_qkv, nullptr, nullptr, qkv,
                                              MROWS, QKVD, DIMX, 0, 0);
  gemm_wmma<<<dim3(32, 36), 128, 0, stream>>>(mask, w_qkv, nullptr, nullptr, qkvm,
                                              MMROWS, QKVD, DIMX, 1, 0);
  colmax_kernel<<<(BATCH * QKVD + 255) / 256, 256, 0, stream>>>(qkvm, cmax);
  updated_kernel<<<(MROWS + 127) / 128, 128, 0, stream>>>(mask, upd);

  // flash attention producing outh (attn@(v*v_m)) and mh (attn@v_m)
  attn_kernel<<<dim3(BATCH * HEADS, 17), 128, 0, stream>>>(qkv, qkvm, cmax, outh, mh);

  // out = (outh*upd) @ w_out^T + b_out ; m = (mh*upd) @ |w_out|^T, drop CLS row
  gemm_wmma<<<dim3(33, 12), 128, 0, stream>>>(outh, w_out, b_out, upd, out0,
                                              MROWS, DIMX, DIMX, 0, 0);
  gemm_wmma<<<dim3(33, 12), 128, 0, stream>>>(mh, w_out, nullptr, upd, out1,
                                              MROWS, DIMX, DIMX, 1, 1);
}
